// IterSpatialCorrelationSampler_62130996904370
// MI455X (gfx1250) — matrix operand order repaired and checked by reading, not compiled, not run on hardware
//
#include <hip/hip_runtime.h>
#include <hip/hip_bf16.h>

// CDNA5 (gfx1250) spatial correlation via banded BF16 WMMA.
// out[b,di,dj,h,w] = sum_c in1[b,c,h,w] * in2[b,c,h+di-4,w+dj-4]
#define BB 8
#define CC 128
#define HH 96
#define WW 128
#define KC 32      // channel chunk per WMMA K-loop step
#define KP 40      // padded K-stride in shorts (80B: 16B-aligned, bank-conflict-free)

typedef __attribute__((ext_vector_type(16))) __bf16 v16bf;
typedef __attribute__((ext_vector_type(8)))  float  v8f;

union Frag16 { v16bf v; uint4 q[2]; };

__device__ __forceinline__ unsigned short f2bf(float f) {
    unsigned int u = __float_as_uint(f);
    u += 0x7fffu + ((u >> 16) & 1u);   // round-to-nearest-even
    return (unsigned short)(u >> 16);
}

__global__ __launch_bounds__(256) void corr_wmma_bf16_kernel(
    const float* __restrict__ in1, const float* __restrict__ in2,
    float* __restrict__ out)
{
    // A tile: 8 h x 16 w x KC c (padded stride KP) ; B tile: 16 rows x 32 cols x KC c
    __shared__ __align__(16) unsigned short sA[8 * 16 * KP];   // 10 KB
    __shared__ __align__(16) unsigned short sB[16 * 32 * KP];  // 40 KB

    const int tid  = threadIdx.x;
    const int lane = tid & 31;
    const int wave = tid >> 5;          // 0..7 -> local h row
    const int w0   = blockIdx.x * 16;   // 8 tiles
    const int h0   = blockIdx.y * 8;    // 12 tiles
    const int b    = blockIdx.z;        // 8

    const int half = lane >> 4;         // K-half selector per ISA fragment layouts
    const int l15  = lane & 15;

    const int baseIn = b * CC * HH * WW;

    // Permanent zero columns 24..31 of sB (second N-tile's unused columns).
    for (int idx = tid; idx < 16 * 8 * KC; idx += 256) {
        int c   = idx & 31;
        int col = 24 + ((idx >> 5) & 7);
        int row = idx >> 8;
        sB[(row * 32 + col) * KP + c] = 0;
    }

    const v8f vzero = {0.f, 0.f, 0.f, 0.f, 0.f, 0.f, 0.f, 0.f};
    v8f acc[9][2];
#pragma unroll
    for (int di = 0; di < 9; ++di) { acc[di][0] = vzero; acc[di][1] = vzero; }

#pragma unroll
    for (int c0 = 0; c0 < CC; c0 += KC) {
        __syncthreads();  // protect LDS from previous iteration's readers (and zero-fill)

        // ---- stage in1 via float4 (w-contiguous, 64B aligned): 8h x 16w x 32c ----
        for (int idx4 = tid; idx4 < (8 * 16 * KC) / 4; idx4 += 256) {
            int w4 = idx4 & 3;              // float4 index along w
            int h  = (idx4 >> 2) & 7;
            int c  = idx4 >> 5;             // 0..31
            const float4 v = *(const float4*)
                (in1 + baseIn + (c0 + c) * HH * WW + (h0 + h) * WW + w0 + w4 * 4);
            unsigned short* p = &sA[(h * 16 + w4 * 4) * KP + c];
            p[0 * KP] = f2bf(v.x);
            p[1 * KP] = f2bf(v.y);
            p[2 * KP] = f2bf(v.z);
            p[3 * KP] = f2bf(v.w);
        }
        // ---- stage in2: rows h0-4..h0+11, padded cols w0-4..w0+19, zero-fill OOB ----
        for (int idx = tid; idx < 16 * 24 * KC; idx += 256) {
            int col = idx % 24;
            int rem = idx / 24;
            int row = rem & 15;
            int c   = rem >> 4;
            int hg  = h0 + row - 4;
            int jg  = w0 + col - 4;
            float v = 0.f;
            if ((unsigned)hg < (unsigned)HH && (unsigned)jg < (unsigned)WW)
                v = in2[baseIn + (c0 + c) * HH * WW + hg * WW + jg];
            sB[(row * 32 + col) * KP + c] = f2bf(v);
        }
        __syncthreads();

        // ---- A fragment (16x32 bf16): lane m=l15; contiguous K = 16*half..+15 ----
        Frag16 fa;
        {
            const unsigned short* pa = &sA[(wave * 16 + l15) * KP + half * 16];
            fa.q[0] = *(const uint4*)(pa);
            fa.q[1] = *(const uint4*)(pa + 8);
        }

#pragma unroll
        for (int di = 0; di < 9; ++di) {
            const int row = wave + di;  // staged row index for in2 h + di - 4

            // ---- B fragment t=0 (cols 0..15): lane n=l15; contiguous K = 16*half..+15 ----
            Frag16 fb0;
            {
                const unsigned short* pb = &sB[(row * 32 + l15) * KP + half * 16];
                fb0.q[0] = *(const uint4*)(pb);
                fb0.q[1] = *(const uint4*)(pb + 8);
            }
            acc[di][0] = __builtin_amdgcn_wmma_f32_16x16x32_bf16(
                false, fa.v, false, fb0.v, (short)0, acc[di][0], false, false);

            // ---- B fragment t=1 (cols 16..31; 24..31 are permanent zeros) ----
            Frag16 fb1;
            {
                const unsigned short* pb = &sB[(row * 32 + 16 + l15) * KP + half * 16];
                fb1.q[0] = *(const uint4*)(pb);
                fb1.q[1] = *(const uint4*)(pb + 8);
            }
            acc[di][1] = __builtin_amdgcn_wmma_f32_16x16x32_bf16(
                false, fa.v, false, fb1.v, (short)0, acc[di][1], false, false);
        }
    }

    // ---- scatter band entries: D VGPR r, lane L -> M = 8*half + r, N = l15 ----
    const int h = h0 + wave;
    float* outw = out + b * (81 * HH * WW) + h * WW + w0;
#pragma unroll
    for (int di = 0; di < 9; ++di) {
#pragma unroll
        for (int t = 0; t < 2; ++t) {
#pragma unroll
            for (int r = 0; r < 8; ++r) {
                const int m   = half * 8 + r;
                const int dji = 16 * t + l15 - m;   // dj index, valid 0..8
                if (0 <= dji && dji <= 8) {
                    outw[di * (9 * HH * WW) + dji * (HH * WW) + m] = acc[di][t][r];
                }
            }
        }
    }
}

extern "C" void kernel_launch(void* const* d_in, const int* in_sizes, int n_in,
                              void* d_out, int out_size, void* d_ws, size_t ws_size,
                              hipStream_t stream) {
    (void)in_sizes; (void)n_in; (void)d_ws; (void)ws_size; (void)out_size;
    const float* in1 = (const float*)d_in[0];
    const float* in2 = (const float*)d_in[1];
    float* out = (float*)d_out;
    dim3 grid(WW / 16, HH / 8, BB);   // (8, 12, 8)
    dim3 block(256);                  // 8 wave32 waves, one per h-row
    corr_wmma_bf16_kernel<<<grid, block, 0, stream>>>(in1, in2, out);
}
